// S4DR_70068096467956
// MI455X (gfx1250) — matrix operand order repaired and checked by reading, not compiled, not run on hardware
//
#include <hip/hip_runtime.h>
#include <hip/hip_bf16.h>

#define Bsz 8
#define Hh  512
#define Nn  64
#define Ls  2048
#define RLEN (Ls + 64)   // reversed kernel + zero tail (causal mask for free)
#define LP   2080        // padded LDS row stride (halves) for u staging

typedef __attribute__((ext_vector_type(16))) _Float16 v16h;
typedef __attribute__((ext_vector_type(8)))  _Float16 v8h;
typedef __attribute__((ext_vector_type(8)))  float    v8f;

// ---------------------------------------------------------------------------
// Kernel 0: per-(h,n) discretization coefficients  Ct = C*(exp(dtA)-1)/A, dtA
// ---------------------------------------------------------------------------
__global__ void s4d_coeffs(const float* __restrict__ C_ri,
                           const float* __restrict__ log_dt,
                           const float* __restrict__ A_ri,
                           float4* __restrict__ ct_da) {
  int idx = blockIdx.x * 256 + threadIdx.x;        // h*Nn + n
  if (idx >= Hh * Nn) return;
  int h = idx >> 6, n = idx & 63;
  float dt  = __expf(log_dt[h]);
  float ar  = A_ri[2 * n], ai = A_ri[2 * n + 1];
  float dar = dt * ar, dai = dt * ai;
  float er  = __expf(dar);
  float s, c;
  __sincosf(dai, &s, &c);
  float em1r = er * c - 1.f, em1i = er * s;        // exp(dtA) - 1
  float cr = C_ri[2 * idx], ci = C_ri[2 * idx + 1];
  float nr = cr * em1r - ci * em1i;                // C * (exp(dtA)-1)
  float ni = cr * em1i + ci * em1r;
  float den = ar * ar + ai * ai;
  float4 o;
  o.x = (nr * ar + ni * ai) / den;                 // Ct.re
  o.y = (ni * ar - nr * ai) / den;                 // Ct.im
  o.z = dar;
  o.w = dai;
  ct_da[idx] = o;
}

// ---------------------------------------------------------------------------
// Kernel 1: k[h,l] = 2*Re sum_n Ct * exp(dtA*l); store REVERSED f16 + 0-tail
// ---------------------------------------------------------------------------
__global__ __launch_bounds__(256) void s4d_kgen(const float4* __restrict__ ct_da,
                                                _Float16* __restrict__ rrev) {
  __shared__ float4 sc[Nn];
  int h = blockIdx.x, tid = threadIdx.x;
  if (tid < Nn) sc[tid] = ct_da[h * Nn + tid];
  __syncthreads();
  int l = blockIdx.y * 256 + tid;
  float lf = (float)l;
  float acc = 0.f;
#pragma unroll 4
  for (int n = 0; n < Nn; ++n) {
    float4 cc = sc[n];
    float mag = __expf(cc.z * lf);
    float s, co;
    __sincosf(cc.w * lf, &s, &co);
    acc += mag * (cc.x * co - cc.y * s);
  }
  rrev[(size_t)h * RLEN + (Ls - 1 - l)] = (_Float16)(2.f * acc);
  if (blockIdx.y == 0 && tid < 64)
    rrev[(size_t)h * RLEN + Ls + tid] = (_Float16)0.f;   // causal zero pad
}

// ---------------------------------------------------------------------------
// Kernel 1b: W (2H x H) f32 -> f16
// ---------------------------------------------------------------------------
__global__ void s4d_wcast(const float* __restrict__ W, _Float16* __restrict__ Wh) {
  int i = blockIdx.x * 256 + threadIdx.x;
  Wh[i] = (_Float16)W[i];
}

// ---------------------------------------------------------------------------
// Kernel 2: causal Toeplitz conv via WMMA, fused +D*u and exact GELU.
// Y_h^T tiles: rows=b (16, 8 live), cols=l (16). A from u (LDS, row 8 = zeros
// so no exec divergence), B from reversed kernel. Double-buffered fragments
// (no WMMA hazard NOPs); wave index readfirstlane'd so tile/chunk loop control
// is scalar (s_cmp/s_cbranch instead of exec-mask loops).
// ---------------------------------------------------------------------------
__global__ __launch_bounds__(256) void s4d_conv(const float* __restrict__ u,
                                                const float* __restrict__ D,
                                                const _Float16* __restrict__ rrev,
                                                _Float16* __restrict__ ygT) {
  __shared__ __align__(16) _Float16 su[(Bsz + 1) * LP];   // +1 zero row
  __shared__ __align__(16) _Float16 sr[RLEN];
  int h = blockIdx.x, band = blockIdx.y, tid = threadIdx.x;

  int Lstage = band * 256 + 288;                   // slack for last K-chunk
  if (Lstage > LP) Lstage = LP;
  for (int b = 0; b < Bsz; ++b) {
    const float* up = u + ((size_t)b * Hh + h) * Ls;
    for (int l = tid; l < Lstage; l += 256)
      su[b * LP + l] = (l < Ls) ? (_Float16)up[l] : (_Float16)0.f;
  }
  for (int l = tid; l < LP; l += 256)
    su[Bsz * LP + l] = (_Float16)0.f;              // zero row for lanes b>=8
  for (int z = tid; z < RLEN; z += 256)
    sr[z] = rrev[(size_t)h * RLEN + z];
  __syncthreads();

  int wave = __builtin_amdgcn_readfirstlane(tid >> 5);   // wave-uniform -> SGPR
  int lane = tid & 31;
  int nl = lane & 15, half = lane >> 4;
  int arow = (nl < Bsz) ? nl : Bsz;                // clamp to zero row
  const _Float16* ap = &su[arow * LP + half * 16];
  float Dh = D[h];

  for (int t = wave; t < 16; t += 8) {             // two 16-col l-tiles / wave
    int l0 = band * 256 + t * 16;                  // scalar
    int zb0 = (Ls - 1) - (l0 + nl) + half * 16;    // reversed-kernel base
    v8f acc = {};
    int nch = (l0 + 16 + 31) >> 5;                 // scalar K-chunk count

    auto loadAB = [&](int j0, v16h& a, v16h& b) {
      v8h r1 = *(const v8h*)(ap + j0);
      v8h r2 = *(const v8h*)(ap + j0 + 8);
      a = __builtin_shufflevector(r1, r2, 0, 1, 2, 3, 4, 5, 6, 7,
                                  8, 9, 10, 11, 12, 13, 14, 15);
      int zb = zb0 + j0;
#pragma unroll
      for (int i = 0; i < 16; ++i) b[i] = sr[zb + i];
    };

    v16h a0, b0, a1, b1;
    loadAB(0, a0, b0);
    int ch = 1;
    for (; ch + 1 < nch; ch += 2) {
      loadAB(ch << 5, a1, b1);
      acc = __builtin_amdgcn_wmma_f32_16x16x32_f16(false, a0, false, b0,
                                                   (short)0, acc, false, false);
      loadAB((ch + 1) << 5, a0, b0);
      acc = __builtin_amdgcn_wmma_f32_16x16x32_f16(false, a1, false, b1,
                                                   (short)0, acc, false, false);
    }
    if (ch < nch) {
      loadAB(ch << 5, a1, b1);
      acc = __builtin_amdgcn_wmma_f32_16x16x32_f16(false, a0, false, b0,
                                                   (short)0, acc, false, false);
      acc = __builtin_amdgcn_wmma_f32_16x16x32_f16(false, a1, false, b1,
                                                   (short)0, acc, false, false);
    } else {
      acc = __builtin_amdgcn_wmma_f32_16x16x32_f16(false, a0, false, b0,
                                                   (short)0, acc, false, false);
    }

    // C/D tile: lanes 0-15 hold VGPR v -> row b=v (b<8 all live on half==0)
    if (half == 0) {
      int l = l0 + nl;
#pragma unroll
      for (int v = 0; v < Bsz; ++v) {
        float x = acc[v] + Dh * (float)su[v * LP + l];
        float g = 0.5f * x * (1.f + erff(x * 0.70710678118f));  // exact GELU
        ygT[((size_t)v * Ls + l) * Hh + h] = (_Float16)g;       // [b][l][h]
      }
    }
  }
}

// ---------------------------------------------------------------------------
// Kernel 3: z = W @ gelu(y) + bias, GLU.  A from W(LDS), B from ygT(global).
// a-half and g-half share every B fragment -> 2 wmma per K-chunk.
// Double-buffered (fixed 16 chunks over H=512), wave index scalarized.
// ---------------------------------------------------------------------------
__global__ __launch_bounds__(256) void s4d_glu(const _Float16* __restrict__ Wh,
                                               const _Float16* __restrict__ ygT,
                                               const float* __restrict__ bias,
                                               float* __restrict__ out) {
  __shared__ __align__(16) _Float16 swa[16 * Hh];
  __shared__ __align__(16) _Float16 swg[16 * Hh];
  int o0 = blockIdx.x << 4, tid = threadIdx.x;
  for (int i = tid; i < 16 * Hh; i += 256) {
    int row = i >> 9, col = i & (Hh - 1);
    swa[i] = Wh[(size_t)(o0 + row) * Hh + col];
    swg[i] = Wh[(size_t)(o0 + 512 + row) * Hh + col];
  }
  __syncthreads();

  int wave = __builtin_amdgcn_readfirstlane(tid >> 5);
  int lane = tid & 31, nl = lane & 15, half = lane >> 4;
  int bl0 = (blockIdx.y * 8 + wave) << 4;          // scalar flat (b,l) tile
  const _Float16* yb = ygT + (size_t)(bl0 + nl) * Hh + half * 8;
  const _Float16* wa = &swa[nl * Hh + half * 8];
  const _Float16* wg = &swg[nl * Hh + half * 8];

  auto loadF = [&](int h0, v16h& aa, v16h& ag, v16h& bb) {
    v8h a1 = *(const v8h*)(wa + h0);
    v8h a2 = *(const v8h*)(wa + h0 + 16);
    aa = __builtin_shufflevector(a1, a2, 0, 1, 2, 3, 4, 5, 6, 7,
                                 8, 9, 10, 11, 12, 13, 14, 15);
    v8h g1 = *(const v8h*)(wg + h0);
    v8h g2 = *(const v8h*)(wg + h0 + 16);
    ag = __builtin_shufflevector(g1, g2, 0, 1, 2, 3, 4, 5, 6, 7,
                                 8, 9, 10, 11, 12, 13, 14, 15);
    const v8h* pb = (const v8h*)(yb + h0);
    bb = __builtin_shufflevector(pb[0], pb[2], 0, 1, 2, 3, 4, 5, 6, 7,
                                 8, 9, 10, 11, 12, 13, 14, 15);
  };

  v8f ca = {}, cg = {};
  v16h aa0, ag0, bb0, aa1, ag1, bb1;
  loadF(0, aa0, ag0, bb0);
#pragma unroll
  for (int h0 = 32; h0 < Hh; h0 += 64) {
    loadF(h0, aa1, ag1, bb1);
    ca = __builtin_amdgcn_wmma_f32_16x16x32_f16(false, aa0, false, bb0,
                                                (short)0, ca, false, false);
    cg = __builtin_amdgcn_wmma_f32_16x16x32_f16(false, ag0, false, bb0,
                                                (short)0, cg, false, false);
    loadF(h0 + 32, aa0, ag0, bb0);
    ca = __builtin_amdgcn_wmma_f32_16x16x32_f16(false, aa1, false, bb1,
                                                (short)0, ca, false, false);
    cg = __builtin_amdgcn_wmma_f32_16x16x32_f16(false, ag1, false, bb1,
                                                (short)0, cg, false, false);
  }
  ca = __builtin_amdgcn_wmma_f32_16x16x32_f16(false, aa0, false, bb0,
                                              (short)0, ca, false, false);
  cg = __builtin_amdgcn_wmma_f32_16x16x32_f16(false, ag0, false, bb0,
                                              (short)0, cg, false, false);

  int b_ = bl0 / Ls;
  int l = (bl0 & (Ls - 1)) + nl;
#pragma unroll
  for (int v = 0; v < 8; ++v) {
    int o = o0 + v + half * 8;
    float a = ca[v] + bias[o];
    float g = cg[v] + bias[o + 512];
    out[((size_t)b_ * Hh + o) * Ls + l] = a * (1.f / (1.f + __expf(-g)));
  }
}

// ---------------------------------------------------------------------------
extern "C" void kernel_launch(void* const* d_in, const int* in_sizes, int n_in,
                              void* d_out, int out_size, void* d_ws, size_t ws_size,
                              hipStream_t stream) {
  const float* u      = (const float*)d_in[0];
  const float* C_ri   = (const float*)d_in[1];
  const float* log_dt = (const float*)d_in[2];
  const float* D      = (const float*)d_in[3];
  const float* W      = (const float*)d_in[4];
  const float* bias   = (const float*)d_in[5];
  const float* A_ri   = (const float*)d_in[6];
  float* out = (float*)d_out;

  // workspace layout (~20.5 MB)
  char* ws = (char*)d_ws;
  float4*   ct_da = (float4*)ws;                                // 512 KB
  _Float16* rrev  = (_Float16*)(ws + (size_t)Hh * Nn * 16);     // 2.06 MB
  _Float16* Wh    = (_Float16*)((char*)rrev + (size_t)Hh * RLEN * 2);   // 1 MB
  _Float16* ygT   = (_Float16*)((char*)Wh + (size_t)2 * Hh * Hh * 2);   // 16 MB

  s4d_coeffs<<<(Hh * Nn + 255) / 256, 256, 0, stream>>>(C_ri, log_dt, A_ri, ct_da);
  s4d_kgen  <<<dim3(Hh, Ls / 256), 256, 0, stream>>>(ct_da, rrev);
  s4d_wcast <<<(2 * Hh * Hh) / 256, 256, 0, stream>>>(W, Wh);
  s4d_conv  <<<dim3(Hh, Ls / 256), 256, 0, stream>>>(u, D, rrev, ygT);
  s4d_glu   <<<dim3(Hh / 16, (Bsz * Ls / 16) / 8), 256, 0, stream>>>(Wh, ygT, bias, out);

  (void)in_sizes; (void)n_in; (void)out_size; (void)ws_size;
}
